// ItemAttentionCF_7988639170816
// MI455X (gfx1250) — compile-verified, matrix operand order checked by reading, not run on hardware
//
#include <hip/hip_runtime.h>
#include <hip/hip_bf16.h>

// ---------------- problem constants ----------------
#define BB      8192
#define LL      200
#define KK      100
#define D_EMB   16
#define N_HID   100
#define N_OUT   16

#define LP      208     // L padded to 13 tiles of 16
#define HP      112     // N_HID padded to 7 tiles of 16
#define KP1     32      // proj K (16) padded to one 16x16x32 WMMA step
#define KP2     128     // attn K (100) padded to 4 WMMA K-steps
#define MT      13      // L tiles
#define NT      7       // H tiles
#define NB      4       // batch rows processed per block in k1

typedef __attribute__((ext_vector_type(16))) _Float16 v16h;
typedef __attribute__((ext_vector_type(8)))  _Float16 v8h;
typedef __attribute__((ext_vector_type(8)))  float    v8f;
typedef __attribute__((ext_vector_type(4)))  int      v4i;
// address-space qualified v4i pointers for the async-LDS builtin
typedef v4i __attribute__((address_space(1)))* v4i_gptr;
typedef v4i __attribute__((address_space(3)))* v4i_lptr;

// gfx1250 async global->LDS path (guarded; falls back to plain copy)
#if defined(__gfx1250__) && __has_builtin(__builtin_amdgcn_global_load_async_to_lds_b128)
#define ASYNC_LDS 1
#else
#define ASYNC_LDS 0
#endif

// LDS layout for k1 (f16 element offsets)
#define OFF_XE   0                               // [LP][KP1] f16
#define OFF_X    (OFF_XE + LP*KP1)               // [LP][KP2] f16
#define OFF_WP   (OFF_X  + LP*KP2)               // [HP][KP1] f16
#define OFF_WA   (OFF_WP + HP*KP1)               // [HP][KP2] f16 (contiguous after WP)
#define F16_ELEMS (OFF_WA + HP*KP2)
// float region follows (element offsets within float*)
#define FOFF_ALIN 0                  // [LP]
#define FOFF_A    (FOFF_ALIN + LP)   // [LP]
#define FOFF_ATTH (FOFF_A + LP)      // [HP]
#define FOFF_PB   (FOFF_ATTH + HP)   // [HP]
#define FOFF_AB   (FOFF_PB + HP)     // [HP]
#define FOFF_RED  (FOFF_AB + HP)     // [16]
#define FOFF_POOL (FOFF_RED + 16)    // [2*HP]
#define F32_ELEMS (FOFF_POOL + 2*HP)

static const size_t K1_LDS_BYTES = (size_t)F16_ELEMS * 2 + (size_t)F32_ELEMS * 4;

// Load a 16x32 f16 tile (rows x K) from LDS, row-major with `stride` elements,
// into the CDNA5 WMMA 16-bit A/B register layout:
//   lane r (0..15) <-> row r ; half = lane>>4 selects K offset +8
//   regs 0..3 hold K = 8*half + 0..7 ; regs 4..7 hold K = 16 + 8*half + 0..7
__device__ __forceinline__ v16h load_tile16(const _Float16* base, int stride, int lane) {
  const int half = lane >> 4;
  const int r    = lane & 15;
  const _Float16* p = base + r * stride + (half << 3);
  v8h lo = *(const v8h*)(p);        // K = 8*half .. 8*half+7
  v8h hi = *(const v8h*)(p + 16);   // K = 16+8*half .. +7
  v16h out;
#pragma unroll
  for (int e = 0; e < 8; ++e) { out[e] = lo[e]; out[e + 8] = hi[e]; }
  return out;
}

// ---------------- k0: pad/convert weights to f16 in ws ----------------
__global__ __launch_bounds__(256) void k0_prep(const float* __restrict__ pw,
                                               const float* __restrict__ aw,
                                               _Float16* __restrict__ wp,
                                               _Float16* __restrict__ wa) {
  int i = blockIdx.x * 256 + threadIdx.x;
  if (i < HP * KP1) {                       // proj_w [100][16] -> [112][32]
    int h = i >> 5, d = i & 31;
    wp[i] = (_Float16)((h < N_HID && d < D_EMB) ? pw[h * D_EMB + d] : 0.f);
  } else if (i < HP * KP1 + HP * KP2) {     // attw_w [100][100] -> [112][128]
    int j = i - HP * KP1;
    int g = j >> 7, h = j & 127;
    wa[j] = (_Float16)((g < N_HID && h < N_HID) ? aw[g * N_HID + h] : 0.f);
  }
}

// ---------------- k1: fused per-row attention -> x1[B][100] ----------------
__global__ __launch_bounds__(256) void k1_attention(
    const float* __restrict__ tr_emb, const float* __restrict__ proj_b,
    const float* __restrict__ attw_b, const float* __restrict__ atth_w,
    const float* __restrict__ atth_b, const int* __restrict__ contexts,
    const _Float16* __restrict__ wp, float* __restrict__ x1_out) {
  extern __shared__ char smem_raw[];
  _Float16* s_f16 = (_Float16*)smem_raw;
  _Float16* s_xe = s_f16 + OFF_XE;
  _Float16* s_x  = s_f16 + OFF_X;
  _Float16* s_wp = s_f16 + OFF_WP;
  _Float16* s_wa = s_f16 + OFF_WA;
  float* s_f32  = (float*)(s_f16 + F16_ELEMS);
  float* s_alin = s_f32 + FOFF_ALIN;
  float* s_a    = s_f32 + FOFF_A;
  float* s_atth = s_f32 + FOFF_ATTH;
  float* s_pb   = s_f32 + FOFF_PB;
  float* s_ab   = s_f32 + FOFF_AB;
  float* s_red  = s_f32 + FOFF_RED;
  float* s_pool = s_f32 + FOFF_POOL;

  const int tid  = threadIdx.x;
  const int wave = tid >> 5;
  const int lane = tid & 31;

  // ---- one-time init: zero pad regions, stage weights/bias vectors ----
  for (int i = tid; i < LP * KP1; i += 256) s_xe[i] = (_Float16)0.f;
  for (int i = tid; i < LP * KP2; i += 256) s_x[i]  = (_Float16)0.f;
  for (int i = tid; i < HP; i += 256) {
    s_atth[i] = (i < N_HID) ? atth_w[i] : 0.f;
    s_pb[i]   = (i < N_HID) ? proj_b[i] : 0.f;
    s_ab[i]   = (i < N_HID) ? attw_b[i] : 0.f;
  }
  // weights: wp then wa are contiguous in ws and in LDS (35840 bytes)
#if ASYNC_LDS
  {
    const char* gsrc = (const char*)wp;
    char* ldst = (char*)s_wp;
    const int wbytes = (HP * KP1 + HP * KP2) * 2;
    for (int off = tid * 16; off < wbytes; off += 256 * 16) {
      __builtin_amdgcn_global_load_async_to_lds_b128(
          (v4i_gptr)(gsrc + off), (v4i_lptr)(ldst + off), 0, 0);
    }
#if __has_builtin(__builtin_amdgcn_s_wait_asynccnt)
    __builtin_amdgcn_s_wait_asynccnt(0);
#else
    asm volatile("s_wait_asynccnt 0" ::: "memory");
#endif
  }
#else
  for (int i = tid; i < HP * KP1 + HP * KP2; i += 256) s_wp[i] = wp[i];
#endif

  const float ab0 = atth_b[0];

  for (int j = 0; j < NB; ++j) {
    const int b = blockIdx.x * NB + j;

    // ---- per-row stage: zero a_lin, gather masked embeddings ----
    for (int i = tid; i < LP; i += 256) s_alin[i] = 0.f;
    for (int i = tid; i < LL * D_EMB; i += 256) {
      int l = i >> 4, d = i & 15;
      int c = contexts[(size_t)b * LL + l];
      float v = 0.f;
      if (c >= 0) v = tr_emb[(size_t)c * D_EMB + d];
      s_xe[l * KP1 + d] = (_Float16)v;
    }
    if (j + 1 < NB && tid < LL)
      __builtin_prefetch(contexts + (size_t)(b + 1) * LL + tid, 0, 1);
    __syncthreads();   // (A) xe/alin ready; prior pooling done before s_x rewrite

    // ---- proj GEMM: x = relu(xe @ Wp^T + pb), 13x7 tiles, one WMMA each ----
    for (int t = wave; t < MT * NT; t += 8) {
      int mt = t / NT, nt = t % NT;
      v16h aa = load_tile16(s_xe + mt * 16 * KP1, KP1, lane);
      v16h bb = load_tile16(s_wp + nt * 16 * KP1, KP1, lane);
      v8f c = {};
      c = __builtin_amdgcn_wmma_f32_16x16x32_f16(false, aa, false, bb,
                                                 (short)0, c, false, false);
      const int half = lane >> 4, r = lane & 15;
      const int col = nt * 16 + r;
      const float bias = s_pb[col];
#pragma unroll
      for (int v = 0; v < 8; ++v) {
        int row = mt * 16 + v + 8 * half;
        float z = c[v] + bias;
        s_x[row * KP2 + col] = (_Float16)(z > 0.f ? z : 0.f);
      }
    }
    __syncthreads();   // (B)

    // ---- attn GEMM: z = relu(x @ Wa^T + ab); fuse a_lin += z * atth_w ----
    for (int t = wave; t < MT * NT; t += 8) {
      int mt = t / NT, nt = t % NT;
      v8f c = {};
#pragma unroll
      for (int k0 = 0; k0 < KP2; k0 += 32) {
        v16h aa = load_tile16(s_x  + mt * 16 * KP2 + k0, KP2, lane);
        v16h bb = load_tile16(s_wa + nt * 16 * KP2 + k0, KP2, lane);
        c = __builtin_amdgcn_wmma_f32_16x16x32_f16(false, aa, false, bb,
                                                   (short)0, c, false, false);
      }
      const int half = lane >> 4, r = lane & 15;
      const int col  = nt * 16 + r;
      const float bias = s_ab[col];
      const float aw   = s_atth[col];
#pragma unroll
      for (int v = 0; v < 8; ++v) {
        int row = mt * 16 + v + 8 * half;
        float z = c[v] + bias;
        z = z > 0.f ? z : 0.f;
        if (row < LL) atomicAdd(&s_alin[row], z * aw);
      }
    }
    __syncthreads();   // (C)

    // ---- faithful masked softmax: e = exp(a - max) * (a == 0) ----
    float al = 0.f;
    if (tid < LL) al = s_alin[tid] + ab0;
    float m = (tid < LL) ? al : -3.402823466e38f;
#pragma unroll
    for (int off = 16; off > 0; off >>= 1) m = fmaxf(m, __shfl_xor(m, off, 32));
    if (lane == 0) s_red[wave] = m;
    __syncthreads();
    float amax = fmaxf(fmaxf(fmaxf(s_red[0], s_red[1]), fmaxf(s_red[2], s_red[3])),
                       fmaxf(fmaxf(s_red[4], s_red[5]), fmaxf(s_red[6], s_red[7])));
    __syncthreads();
    float e = (tid < LL) ? (expf(al - amax) * (al == 0.f ? 1.f : 0.f)) : 0.f;
    float ssum = e;
#pragma unroll
    for (int off = 16; off > 0; off >>= 1) ssum += __shfl_xor(ssum, off, 32);
    if (lane == 0) s_red[wave] = ssum;
    __syncthreads();
    float tot = s_red[0] + s_red[1] + s_red[2] + s_red[3] +
                s_red[4] + s_red[5] + s_red[6] + s_red[7];
    if (tid < LL) s_a[tid] = e / (tot + 1e-8f);
    __syncthreads();   // (D)

    // ---- pooled x1[h] = sum_l a[l] * x[l][h]  (2 L-segments in parallel) ----
    {
      const int h   = tid & 127;
      const int seg = tid >> 7;
      if (h < N_HID) {
        float acc = 0.f;
        const int l0 = seg * 100;
        for (int l = l0; l < l0 + 100; ++l)
          acc += s_a[l] * (float)s_x[l * KP2 + h];
        s_pool[seg * HP + h] = acc;
      }
      __syncthreads();
      if (tid < N_HID)
        x1_out[(size_t)b * N_HID + tid] = s_pool[tid] + s_pool[HP + tid];
    }
  }
}

// ---------------- k2: batch-norm stats -> folded scale/shift ----------------
__global__ __launch_bounds__(256) void k2_bnstats(const float* __restrict__ x1,
                                                  const float* __restrict__ gamma,
                                                  const float* __restrict__ beta,
                                                  float* __restrict__ ss) {
  const int h = blockIdx.x;
  const int t = threadIdx.x;
  float s = 0.f, s2 = 0.f;
  for (int b = t; b < BB; b += 256) {
    float v = x1[(size_t)b * N_HID + h];
    s += v; s2 += v * v;
  }
  __shared__ float r1[8], r2[8];
  const int wave = t >> 5, lane = t & 31;
#pragma unroll
  for (int off = 16; off > 0; off >>= 1) {
    s  += __shfl_xor(s,  off, 32);
    s2 += __shfl_xor(s2, off, 32);
  }
  if (lane == 0) { r1[wave] = s; r2[wave] = s2; }
  __syncthreads();
  if (t == 0) {
    float ts = 0.f, ts2 = 0.f;
#pragma unroll
    for (int w = 0; w < 8; ++w) { ts += r1[w]; ts2 += r2[w]; }
    float mean = ts * (1.f / BB);
    float var  = ts2 * (1.f / BB) - mean * mean;   // biased variance
    float rstd = rsqrtf(var + 1e-5f);
    float sc = gamma[h] * rstd;
    ss[h]          = sc;
    ss[N_HID + h]  = beta[h] - mean * sc;
  }
}

// ---------------- k3: BN apply + pl GEMM + out_emb gather-dot ----------------
#define RB 32   // batch rows per block
__global__ __launch_bounds__(256) void k3_out(const float* __restrict__ x1,
                                              const float* __restrict__ ss,
                                              const float* __restrict__ pl_w,
                                              const float* __restrict__ pl_b,
                                              const float* __restrict__ out_emb,
                                              const int* __restrict__ tidx,
                                              float* __restrict__ pred) {
  __shared__ float s_x1[RB][104];
  __shared__ float s_w[N_OUT][104];
  __shared__ float s_xo[RB][N_OUT];
  const int t  = threadIdx.x;
  const int b0 = blockIdx.x * RB;

  for (int i = t; i < N_OUT * N_HID; i += 256) {
    int o = i / N_HID, h = i % N_HID;
    s_w[o][h] = pl_w[o * N_HID + h];
  }
  for (int i = t; i < RB * N_HID; i += 256) {
    int r = i / N_HID, h = i % N_HID;
    s_x1[r][h] = x1[(size_t)(b0 + r) * N_HID + h] * ss[h] + ss[N_HID + h];
  }
  __syncthreads();
  for (int i = t; i < RB * N_OUT; i += 256) {
    int r = i >> 4, o = i & 15;
    float acc = pl_b[o];
    for (int h = 0; h < N_HID; ++h) acc += s_x1[r][h] * s_w[o][h];
    s_xo[r][o] = acc;
  }
  __syncthreads();
  for (int i = t; i < RB * KK; i += 256) {
    int r = i / KK, k = i % KK;
    int b = b0 + r;
    int it = tidx[(size_t)b * KK + k];
    const float4* yp = (const float4*)(out_emb + (size_t)it * N_OUT);
    float4 y0 = yp[0], y1 = yp[1], y2 = yp[2], y3 = yp[3];
    const float* xo = s_xo[r];
    float acc = y0.x * xo[0]  + y0.y * xo[1]  + y0.z * xo[2]  + y0.w * xo[3]
              + y1.x * xo[4]  + y1.y * xo[5]  + y1.z * xo[6]  + y1.w * xo[7]
              + y2.x * xo[8]  + y2.y * xo[9]  + y2.z * xo[10] + y2.w * xo[11]
              + y3.x * xo[12] + y3.y * xo[13] + y3.z * xo[14] + y3.w * xo[15];
    pred[(size_t)b * KK + k] = acc;
  }
}

// ---------------- launcher ----------------
extern "C" void kernel_launch(void* const* d_in, const int* in_sizes, int n_in,
                              void* d_out, int out_size, void* d_ws, size_t ws_size,
                              hipStream_t stream) {
  const float* tr_emb   = (const float*)d_in[0];
  const float* out_emb  = (const float*)d_in[1];
  const float* proj_w   = (const float*)d_in[2];
  const float* proj_b   = (const float*)d_in[3];
  const float* attw_w   = (const float*)d_in[4];
  const float* attw_b   = (const float*)d_in[5];
  const float* atth_w   = (const float*)d_in[6];
  const float* atth_b   = (const float*)d_in[7];
  const float* bn_gamma = (const float*)d_in[8];
  const float* bn_beta  = (const float*)d_in[9];
  const float* pl_w     = (const float*)d_in[10];
  const float* pl_b     = (const float*)d_in[11];
  // d_in[12] = pid (unused by reference)
  const int*   tidx     = (const int*)d_in[13];
  const int*   contexts = (const int*)d_in[14];
  float* pred = (float*)d_out;

  // workspace layout (wp and wa contiguous)
  char* ws = (char*)d_ws;
  _Float16* wp = (_Float16*)ws;                               // 112*32  f16
  _Float16* wa = (_Float16*)(ws + HP * KP1 * 2);              // 112*128 f16
  float* x1 = (float*)(ws + HP * KP1 * 2 + HP * KP2 * 2);     // 8192*100 f32
  float* ss = x1 + (size_t)BB * N_HID;                        // 200 f32

  // allow >64KB dynamic LDS for k1
  (void)hipFuncSetAttribute((const void*)k1_attention,
                            hipFuncAttributeMaxDynamicSharedMemorySize,
                            (int)K1_LDS_BYTES);

  int prep_elems = HP * KP1 + HP * KP2;
  k0_prep<<<(prep_elems + 255) / 256, 256, 0, stream>>>(proj_w, attw_w, wp, wa);
  k1_attention<<<BB / NB, 256, K1_LDS_BYTES, stream>>>(tr_emb, proj_b, attw_b,
                                                       atth_w, atth_b, contexts,
                                                       wp, x1);
  k2_bnstats<<<N_HID, 256, 0, stream>>>(x1, bn_gamma, bn_beta, ss);
  k3_out<<<BB / RB, 256, 0, stream>>>(x1, ss, pl_w, pl_b, out_emb, tidx, pred);
}
